// ViT_72541997629946
// MI455X (gfx1250) — matrix operand order repaired and checked by reading, not compiled, not run on hardware
//
#include <hip/hip_runtime.h>

typedef __attribute__((ext_vector_type(16))) _Float16 v16h;
typedef __attribute__((ext_vector_type(8)))  _Float16 v8h;
typedef __attribute__((ext_vector_type(8)))  float    v8f;

#define TM 128
#define TN 128
#define TK 32
#define TKP 40   // padded LDS row stride (halfs): 80B -> 16B aligned, bank-conflict free

// ---------------------------------------------------------------------------
// CDNA5 async global->LDS copy (16B per lane), tracked with ASYNCcnt.
// ---------------------------------------------------------------------------
__device__ __forceinline__ void async_copy16(void* lds, const void* gptr)
{
    unsigned l = (unsigned)(uintptr_t)lds;   // low 32 bits = LDS byte offset
    asm volatile("global_load_async_to_lds_b128 %0, %1, off"
                 :: "v"(l), "v"(gptr) : "memory");
}
__device__ __forceinline__ void wait_async0()
{
    asm volatile("s_wait_asynccnt 0x0" ::: "memory");
}

// ---------------------------------------------------------------------------
// Generic batched GEMM, f16 inputs, f16 WMMA compute, f32 accumulate,
// optional f32 and/or f16 outputs, fused epilogue.
//   C[m][n] = epi( alpha * sum_k A[m][k] * B'[k][n] )
//   B'[k][n] = transB ? Bmat[n*ldb + k] : Bmat[k*ldb + n]
//   epi: 0 = none, 1 = +bias, 2 = gelu(.+bias), 3 = residual + (.+bias)
// Per-batch bases: base = (bb/bmod)*s1 + (bb%bmod)*s2
// REQUIREMENTS (caller guarantees):
//   - K is a multiple of 32; rows of A (and of B when transB) are readable for
//     the full K span (zero-padded where k >= logical K).
//   - Out-of-tile rows/cols are handled by index clamping (their products feed
//     only unstored outputs), so buffers need no other guards.
//   - For transB==0, Bmat must be readable at [k][n0..n0+127] for all staged k
//     (tail slack in the buffer).
// ---------------------------------------------------------------------------
__global__ __launch_bounds__(256) void vit_gemm_wmma(
    const _Float16* __restrict__ A, const _Float16* __restrict__ Bmat,
    const float* __restrict__ bias, const float* __restrict__ Res,
    float* __restrict__ C, _Float16* __restrict__ Ch,
    int M, int N, int K,
    int lda, int ldb, int ldc,
    int bmod,
    long long as1, long long as2, long long bs1, long long bs2,
    long long cs1, long long cs2,
    int transB, int epi, float alpha)
{
    __shared__ __align__(16) _Float16 As[2][TM][TKP];
    __shared__ __align__(16) _Float16 Bs[2][TN][TKP];

    const int bb = blockIdx.z;
    const long long abase = (long long)(bb / bmod) * as1 + (long long)(bb % bmod) * as2;
    const long long bbase = (long long)(bb / bmod) * bs1 + (long long)(bb % bmod) * bs2;
    const long long cbase = (long long)(bb / bmod) * cs1 + (long long)(bb % bmod) * cs2;

    const int m0 = blockIdx.y * TM;
    const int n0 = blockIdx.x * TN;

    const int tid  = threadIdx.x;
    const int lane = tid & 31;
    const int wave = tid >> 5;     // 0..7
    const int wm   = wave & 3;     // 32-row chunk within 128
    const int wn   = wave >> 2;    // 64-col chunk within 128
    const int l16  = lane & 15;
    const int hlf  = lane >> 4;    // 0 / 1

    // staging coordinates (row-clamped; no per-element guards)
    const int sa_m  = tid >> 1;
    const int sa_kp = (tid & 1) * 16;
    const int sa_gm = (m0 + sa_m < M) ? (m0 + sa_m) : (M - 1);
    const _Float16* sa_p = A + abase + (long long)sa_gm * lda + sa_kp;

    const int sb_n  = tid >> 1;
    const int sb_kp = (tid & 1) * 16;
    const int sb_gn = (n0 + sb_n < N) ? (n0 + sb_n) : (N - 1);
    const _Float16* sb_p = Bmat + bbase + (long long)sb_gn * ldb + sb_kp;  // transB

    const int sv_k  = tid >> 3;            // 0..31
    const int sv_nb = (tid & 7) * 16;      // 0..112
    const _Float16* sv_p = Bmat + bbase + (long long)sv_k * ldb + n0 + sv_nb; // !transB

    v8f acc[2][4];
#pragma unroll
    for (int i = 0; i < 2; i++)
#pragma unroll
        for (int j = 0; j < 4; j++)
#pragma unroll
            for (int e = 0; e < 8; e++) acc[i][j][e] = 0.0f;

    const int nsteps = K / TK;

    // ---- stage first tile into buffer 0 ----
    {
        async_copy16(&As[0][sa_m][sa_kp],     sa_p);
        async_copy16(&As[0][sa_m][sa_kp + 8], sa_p + 8);
        if (transB) {
            async_copy16(&Bs[0][sb_n][sb_kp],     sb_p);
            async_copy16(&Bs[0][sb_n][sb_kp + 8], sb_p + 8);
        } else {
            v8h x0 = *(const v8h*)(sv_p);
            v8h x1 = *(const v8h*)(sv_p + 8);
#pragma unroll
            for (int i = 0; i < 8; i++) {
                Bs[0][sv_nb + i][sv_k]     = x0[i];
                Bs[0][sv_nb + 8 + i][sv_k] = x1[i];
            }
        }
    }
    wait_async0();
    __syncthreads();

    for (int s = 0; s < nsteps; s++) {
        const int cur = s & 1;
        const int nxt = cur ^ 1;

        // ---- prefetch next tile into the other buffer ----
        if (s + 1 < nsteps) {
            const int k0 = (s + 1) * TK;
            async_copy16(&As[nxt][sa_m][sa_kp],     sa_p + k0);
            async_copy16(&As[nxt][sa_m][sa_kp + 8], sa_p + k0 + 8);
            if (transB) {
                async_copy16(&Bs[nxt][sb_n][sb_kp],     sb_p + k0);
                async_copy16(&Bs[nxt][sb_n][sb_kp + 8], sb_p + k0 + 8);
            } else {
                const _Float16* bp = sv_p + (long long)k0 * ldb;
                v8h x0 = *(const v8h*)(bp);
                v8h x1 = *(const v8h*)(bp + 8);
#pragma unroll
                for (int i = 0; i < 8; i++) {
                    Bs[nxt][sv_nb + i][sv_k]     = x0[i];
                    Bs[nxt][sv_nb + 8 + i][sv_k] = x1[i];
                }
            }
        }

        // ---- fragments per ISA VGPR layouts ----
        // A 16x32 f16: lanes0-15 K=0..7 & 16..23; lanes16-31 K=8..15 & 24..31
        v16h afrag[2];
#pragma unroll
        for (int mi = 0; mi < 2; mi++) {
            const int m  = wm * 32 + mi * 16 + l16;
            const int kb = hlf * 8;
            v8h lo = *(const v8h*)&As[cur][m][kb];
            v8h hi = *(const v8h*)&As[cur][m][kb + 16];
            v16h f;
#pragma unroll
            for (int e = 0; e < 8; e++) { f[e] = lo[e]; f[e + 8] = hi[e]; }
            afrag[mi] = f;
        }
        // B 32x16 f16: lanes0-15 K=0..7 & 16..23; lanes16-31 K=8..15 & 24..31 (per column n)
        v16h bfrag[4];
#pragma unroll
        for (int ni = 0; ni < 4; ni++) {
            const int n  = wn * 64 + ni * 16 + l16;
            const int kb = hlf * 8;
            v8h lo = *(const v8h*)&Bs[cur][n][kb];
            v8h hi = *(const v8h*)&Bs[cur][n][kb + 16];
            v16h f;
#pragma unroll
            for (int e = 0; e < 8; e++) { f[e] = lo[e]; f[e + 8] = hi[e]; }
            bfrag[ni] = f;
        }
#pragma unroll
        for (int mi = 0; mi < 2; mi++)
#pragma unroll
            for (int ni = 0; ni < 4; ni++)
                acc[mi][ni] = __builtin_amdgcn_wmma_f32_16x16x32_f16(
                    false, afrag[mi], false, bfrag[ni],
                    (short)0, acc[mi][ni], false, false);

        wait_async0();      // next tile fully in LDS (this wave's copies)
        __syncthreads();    // visible to all waves
    }

    // ---- epilogue + store (C/D layout: VGPR r -> M = r + hlf*8, N = l16) ----
#pragma unroll
    for (int mi = 0; mi < 2; mi++)
#pragma unroll
        for (int ni = 0; ni < 4; ni++)
#pragma unroll
            for (int r = 0; r < 8; r++) {
                const int m = m0 + wm * 32 + mi * 16 + hlf * 8 + r;
                const int n = n0 + wn * 64 + ni * 16 + l16;
                if (m < M && n < N) {
                    float v = acc[mi][ni][r] * alpha;
                    if (epi >= 1 && bias) v += bias[n];
                    if (epi == 2) {
                        const float t = v;
                        v = 0.5f * t * (1.0f + tanhf(0.7978845608f * (t + 0.044715f * t * t * t)));
                    }
                    const long long cidx = cbase + (long long)m * ldc + n;
                    if (epi == 3 && Res) v += Res[cidx];
                    if (C)  C[cidx]  = v;
                    if (Ch) Ch[cidx] = (_Float16)v;
                }
            }
}

// ---------------------------------------------------------------------------
// LDS-tiled transpose + f32->f16 convert: in[k][n] -> out[n][k]
// ---------------------------------------------------------------------------
__global__ __launch_bounds__(256) void vit_transpose_cvt(
    const float* __restrict__ in, _Float16* __restrict__ out, int K, int N)
{
    __shared__ float tile[32][33];
    const int k0 = blockIdx.y * 32, n0 = blockIdx.x * 32;
    const int tx = threadIdx.x, ty = threadIdx.y;   // 32 x 8
#pragma unroll
    for (int i = 0; i < 32; i += 8) {
        const int k = k0 + ty + i, n = n0 + tx;
        tile[ty + i][tx] = (k < K && n < N) ? in[(long long)k * N + n] : 0.0f;
    }
    __syncthreads();
#pragma unroll
    for (int i = 0; i < 32; i += 8) {
        const int n = n0 + ty + i, k = k0 + tx;
        if (n < N && k < K) out[(long long)n * K + k] = (_Float16)tile[tx][ty + i];
    }
}

// ---------------------------------------------------------------------------
// Wave-per-row LayerNorm (population variance, eps 1e-5) -> f16 output
// ---------------------------------------------------------------------------
__global__ __launch_bounds__(256) void vit_layernorm(
    const float* __restrict__ X, _Float16* __restrict__ Y,
    const float* __restrict__ g, const float* __restrict__ b,
    int rows, int width, long long in_stride, long long out_stride)
{
    const int row  = blockIdx.x * 8 + (threadIdx.x >> 5);
    const int lane = threadIdx.x & 31;
    if (row >= rows) return;
    const float* x = X + (long long)row * in_stride;
    float s = 0.0f, s2 = 0.0f;
    for (int i = lane; i < width; i += 32) { const float v = x[i]; s += v; s2 += v * v; }
    for (int off = 16; off > 0; off >>= 1) {
        s  += __shfl_xor(s, off, 32);
        s2 += __shfl_xor(s2, off, 32);
    }
    const float mean = s / (float)width;
    const float var  = s2 / (float)width - mean * mean;
    const float inv  = rsqrtf(var + 1e-5f);
    _Float16* y = Y + (long long)row * out_stride;
    for (int i = lane; i < width; i += 32)
        y[i] = (_Float16)((x[i] - mean) * inv * g[i] + b[i]);
}

// ---------------------------------------------------------------------------
// Wave-per-row softmax: f32 scores (width 197) -> f16 probs, row stride 224,
// zero-padded tail (cols 197..223) so the P@V GEMM can run K=224 guard-free.
// ---------------------------------------------------------------------------
__global__ __launch_bounds__(256) void vit_softmax(
    const float* __restrict__ S, _Float16* __restrict__ P, int rows)
{
    const int row  = blockIdx.x * 8 + (threadIdx.x >> 5);
    const int lane = threadIdx.x & 31;
    if (row >= rows) return;
    const float* p = S + (long long)row * 197;
    _Float16*    q = P + (long long)row * 224;
    float mx = -1e30f;
    for (int i = lane; i < 197; i += 32) mx = fmaxf(mx, p[i]);
    for (int off = 16; off > 0; off >>= 1) mx = fmaxf(mx, __shfl_xor(mx, off, 32));
    float sum = 0.0f;
    float e[7];
    for (int i = lane, j = 0; i < 197; i += 32, j++) { e[j] = __expf(p[i] - mx); sum += e[j]; }
    for (int off = 16; off > 0; off >>= 1) sum += __shfl_xor(sum, off, 32);
    const float r = 1.0f / sum;
    for (int i = lane, j = 0; i < 197; i += 32, j++) q[i] = (_Float16)(e[j] * r);
    for (int i = 197 + lane; i < 224; i += 32) q[i] = (_Float16)0.0f;
}

// ---------------------------------------------------------------------------
// img [B,3,224,224] -> f16 patches [B*196, 768] (col = pr*48 + pc*3 + c)
// ---------------------------------------------------------------------------
__global__ void vit_patchify(const float* __restrict__ img, _Float16* __restrict__ P)
{
    const long long idx = (long long)blockIdx.x * blockDim.x + threadIdx.x;
    const long long total = 32LL * 196 * 768;
    if (idx >= total) return;
    const int col = (int)(idx % 768);
    const long long row = idx / 768;
    const int b  = (int)(row / 196);
    const int p  = (int)(row % 196);
    const int py = p / 14, px = p % 14;
    const int c  = col % 3;
    const int pc = (col / 3) % 16;
    const int pr = col / 48;
    const int hh = py * 16 + pr, ww = px * 16 + pc;
    P[idx] = (_Float16)img[(((long long)(b * 3 + c)) * 224 + hh) * 224 + ww];
}

// ---------------------------------------------------------------------------
// X[b,0,:] = cls + pos[0]; X[b,1+t,:] = emb[b,t,:] + pos[1+t]   (f32)
// ---------------------------------------------------------------------------
__global__ void vit_assemble(const float* __restrict__ emb, const float* __restrict__ cls,
                             const float* __restrict__ pos, float* __restrict__ X)
{
    const long long idx = (long long)blockIdx.x * blockDim.x + threadIdx.x;
    const long long total = 32LL * 197 * 768;
    if (idx >= total) return;
    const int n = (int)(idx % 768);
    const long long bt = idx / 768;
    const int t = (int)(bt % 197);
    const int b = (int)(bt / 197);
    const float v = (t == 0) ? cls[n] : emb[((long long)b * 196 + (t - 1)) * 768 + n];
    X[idx] = v + pos[(long long)t * 768 + n];
}

// ---------------------------------------------------------------------------
// Host side
// ---------------------------------------------------------------------------
static inline void launch_gemm(hipStream_t st,
                               const _Float16* A, const _Float16* Bm,
                               const float* bias, const float* Res,
                               float* C, _Float16* Ch,
                               int M, int N, int K,
                               int lda, int ldb, int ldc,
                               int batch, int bmod,
                               long long as1, long long as2,
                               long long bs1, long long bs2,
                               long long cs1, long long cs2,
                               int transB, int epi, float alpha)
{
    dim3 grid((N + TN - 1) / TN, (M + TM - 1) / TM, batch);
    vit_gemm_wmma<<<grid, 256, 0, st>>>(A, Bm, bias, Res, C, Ch,
                                        M, N, K, lda, ldb, ldc,
                                        bmod, as1, as2, bs1, bs2, cs1, cs2,
                                        transB, epi, alpha);
}

static inline void launch_transpose(hipStream_t st, const float* in, _Float16* out,
                                    int K, int N)
{
    dim3 grid((N + 31) / 32, (K + 31) / 32);
    vit_transpose_cvt<<<grid, dim3(32, 8), 0, st>>>(in, out, K, N);
}

extern "C" void kernel_launch(void* const* d_in, const int* in_sizes, int n_in,
                              void* d_out, int out_size, void* d_ws, size_t ws_size,
                              hipStream_t stream)
{
    (void)in_sizes; (void)n_in; (void)out_size; (void)ws_size;

    const float* img    = (const float*)d_in[0];
    const float* pe_w   = (const float*)d_in[1];
    const float* pe_b   = (const float*)d_in[2];
    const float* pos    = (const float*)d_in[3];
    const float* cls    = (const float*)d_in[4];
    const float* ln1_g  = (const float*)d_in[5];
    const float* ln1_b  = (const float*)d_in[6];
    const float* wq     = (const float*)d_in[7];
    const float* wk     = (const float*)d_in[8];
    const float* wv     = (const float*)d_in[9];
    const float* wo     = (const float*)d_in[10];
    const float* bo     = (const float*)d_in[11];
    const float* ln2_g  = (const float*)d_in[12];
    const float* ln2_b  = (const float*)d_in[13];
    const float* w1     = (const float*)d_in[14];
    const float* b1     = (const float*)d_in[15];
    const float* w2     = (const float*)d_in[16];
    const float* b2     = (const float*)d_in[17];
    const float* lnf_g  = (const float*)d_in[18];
    const float* lnf_b  = (const float*)d_in[19];
    const float* head_w = (const float*)d_in[20];
    const float* head_b = (const float*)d_in[21];
    float* out = (float*)d_out;

    const int Bn = 32, T = 197, Dm = 768, NH = 12, DH = 64, MLPD = 3072, NCLS = 1000;
    const int Rows = Bn * T;                                  // 6304
    const long long nX   = (long long)Rows * Dm;              // 4,841,472
    const long long nS   = (long long)Bn * NH * T * T;        // 14,902,656
    const long long nP   = (long long)Bn * NH * T * 224;      // 16,945,152
    const long long nU   = (long long)Rows * MLPD;            // 19,365,888
    const long long nWd  = (long long)Dm * Dm;                // 589,824
    const long long nW1  = (long long)Dm * MLPD;              // 2,359,296
    const long long slack = 256LL * Dm;                       // tail slack for clamp-free reads

    // ---- workspace layout (fp32 first, then f16; all 16B aligned) ----
    float* X = (float*)d_ws;
    float* S = X + nX;                       // scores; also emb f32 temp pre-layer0
    _Float16* h16 = (_Float16*)(S + nS);
    _Float16* Hh = h16;            h16 += nX + slack;
    _Float16* Qh = h16;            h16 += nX + slack;
    _Float16* Kh = h16;            h16 += nX + slack;
    _Float16* Vh = h16;            h16 += nX + slack;
    _Float16* Oh = h16;            h16 += nX + slack;
    _Float16* Ph = h16;            h16 += nP + slack;
    _Float16* Uh = h16;            h16 += nU + slack;   // also f16 patch matrix
    _Float16* WqT = h16;           h16 += 12 * nWd;
    _Float16* WkT = h16;           h16 += 12 * nWd;
    _Float16* WvT = h16;           h16 += 12 * nWd;
    _Float16* WoT = h16;           h16 += 12 * nWd;
    _Float16* W1T = h16;           h16 += 12 * nW1;
    _Float16* W2T = h16;           h16 += 12 * nW1;
    _Float16* PeT = h16;           h16 += nWd;
    _Float16* HdT = h16;           h16 += (long long)NCLS * Dm;

    // ---- one-time weight transpose + f16 conversion ----
    launch_transpose(stream, pe_w, PeT, Dm, Dm);
    launch_transpose(stream, head_w, HdT, Dm, NCLS);
    for (int l = 0; l < 12; l++) {
        launch_transpose(stream, wq + (long long)l * nWd, WqT + (long long)l * nWd, Dm, Dm);
        launch_transpose(stream, wk + (long long)l * nWd, WkT + (long long)l * nWd, Dm, Dm);
        launch_transpose(stream, wv + (long long)l * nWd, WvT + (long long)l * nWd, Dm, Dm);
        launch_transpose(stream, wo + (long long)l * nWd, WoT + (long long)l * nWd, Dm, Dm);
        launch_transpose(stream, w1 + (long long)l * nW1, W1T + (long long)l * nW1, Dm, MLPD);
        launch_transpose(stream, w2 + (long long)l * nW1, W2T + (long long)l * nW1, MLPD, Dm);
    }

    // ---- patch embedding ----
    {
        const long long np_el = 32LL * 196 * 768;
        vit_patchify<<<(unsigned)((np_el + 255) / 256), 256, 0, stream>>>(img, Uh);
        // emb(f32, in S temp) = patches @ pe_w + pe_b
        launch_gemm(stream, Uh, PeT, pe_b, nullptr, S, nullptr,
                    Bn * 196, Dm, Dm, Dm, Dm, Dm,
                    1, 1, 0,0, 0,0, 0,0, 1, 1, 1.0f);
        const long long nx_el = (long long)Rows * Dm;
        vit_assemble<<<(unsigned)((nx_el + 255) / 256), 256, 0, stream>>>(S, cls, pos, X);
    }

    const float scale = 0.125f;                        // DH^-0.5
    const long long sBT = (long long)T * Dm;           // per-image stride in QKV
    const long long sSc = (long long)T * T;            // per-(b,h) f32 score stride
    const long long sPr = (long long)T * 224;          // per-(b,h) f16 prob stride

    for (int l = 0; l < 12; l++) {
        const _Float16* lWq = WqT + (long long)l * nWd;
        const _Float16* lWk = WkT + (long long)l * nWd;
        const _Float16* lWv = WvT + (long long)l * nWd;
        const _Float16* lWo = WoT + (long long)l * nWd;
        const _Float16* lW1 = W1T + (long long)l * nW1;
        const _Float16* lW2 = W2T + (long long)l * nW1;
        const float* Bo = bo + (long long)l * Dm;
        const float* B1 = b1 + (long long)l * MLPD;
        const float* B2 = b2 + (long long)l * Dm;

        // LN1: X -> Hh (f16)
        vit_layernorm<<<(Rows + 7) / 8, 256, 0, stream>>>(
            X, Hh, ln1_g + l * Dm, ln1_b + l * Dm, Rows, Dm, Dm, Dm);

        // QKV (f16 out only)
        launch_gemm(stream, Hh, lWq, nullptr, nullptr, nullptr, Qh,
                    Rows, Dm, Dm, Dm, Dm, Dm,
                    1, 1, 0,0, 0,0, 0,0, 1, 0, 1.0f);
        launch_gemm(stream, Hh, lWk, nullptr, nullptr, nullptr, Kh,
                    Rows, Dm, Dm, Dm, Dm, Dm,
                    1, 1, 0,0, 0,0, 0,0, 1, 0, 1.0f);
        launch_gemm(stream, Hh, lWv, nullptr, nullptr, nullptr, Vh,
                    Rows, Dm, Dm, Dm, Dm, Dm,
                    1, 1, 0,0, 0,0, 0,0, 1, 0, 1.0f);

        // scores(f32) = scale * q @ k^T  (batched over B*NH; base = b*sBT + h*64)
        launch_gemm(stream, Qh, Kh, nullptr, nullptr, S, nullptr,
                    T, T, DH, Dm, Dm, T,
                    Bn * NH, NH,
                    sBT, 64, sBT, 64, (long long)NH * sSc, sSc,
                    1, 0, scale);

        // softmax -> f16 probs (stride 224, zero tail)
        vit_softmax<<<(Bn * NH * T + 7) / 8, 256, 0, stream>>>(S, Ph, Bn * NH * T);

        // O(f16) = P @ v  -> Oh as [B,T,H*DH]; K padded to 224 (zeros in P)
        launch_gemm(stream, Ph, Vh, nullptr, nullptr, nullptr, Oh,
                    T, DH, 224, 224, Dm, Dm,
                    Bn * NH, NH,
                    (long long)NH * sPr, sPr, sBT, 64, sBT, 64,
                    0, 0, 1.0f);

        // X = X + O @ Wo + Bo
        launch_gemm(stream, Oh, lWo, Bo, X, X, nullptr,
                    Rows, Dm, Dm, Dm, Dm, Dm,
                    1, 1, 0,0, 0,0, 0,0, 1, 3, 1.0f);

        // LN2: X -> Hh
        vit_layernorm<<<(Rows + 7) / 8, 256, 0, stream>>>(
            X, Hh, ln2_g + l * Dm, ln2_b + l * Dm, Rows, Dm, Dm, Dm);

        // Uh = gelu(Hh @ W1 + B1)   (f16 out only)
        launch_gemm(stream, Hh, lW1, B1, nullptr, nullptr, Uh,
                    Rows, MLPD, Dm, Dm, Dm, MLPD,
                    1, 1, 0,0, 0,0, 0,0, 1, 2, 1.0f);

        // X = X + Uh @ W2 + B2
        launch_gemm(stream, Uh, lW2, B2, X, X, nullptr,
                    Rows, Dm, MLPD, MLPD, MLPD, Dm,
                    1, 1, 0,0, 0,0, 0,0, 1, 3, 1.0f);
    }

    // pooled = LN(X[:,0,:]) -> Hh (32 rows)
    vit_layernorm<<<(Bn + 7) / 8, 256, 0, stream>>>(
        X, Hh, lnf_g, lnf_b, Bn, Dm, (long long)T * Dm, Dm);

    // out = pooled @ head_w + head_b
    launch_gemm(stream, Hh, HdT, head_b, nullptr, out, nullptr,
                Bn, NCLS, Dm, Dm, Dm, NCLS,
                1, 1, 0,0, 0,0, 0,0, 1, 1, 1.0f);
}
